// PABBlock_40724879901258
// MI455X (gfx1250) — matrix operand order repaired and checked by reading, not compiled, not run on hardware
//
#include <hip/hip_runtime.h>
#include <hip/hip_bf16.h>
#include <math.h>

// ---------------------------------------------------------------------------
// PAB block on gfx1250: all GEMMs via v_wmma_f32_16x16x32_bf16 (wave32),
// flash-style global softmax. Fragments load as contiguous 16B chunks
// (ds_load_b128 / global_load_b128); attention tiles stream into LDS with
// global_load_async_to_lds_b128 (ASYNCcnt).
// ---------------------------------------------------------------------------

typedef __bf16 bf16_t;
typedef __attribute__((ext_vector_type(16))) __bf16 v16bf;
typedef __attribute__((ext_vector_type(8)))  __bf16 v8bf;
typedef __attribute__((ext_vector_type(8)))  float  v8f;

#define B_  8
#define C_  256
#define P_  64
#define HW_ 4096
#define K9_ 2304   // C_ * 9 taps

static __device__ __forceinline__ v8f wmma_bf16(v16bf a, v16bf b, v8f c) {
  return __builtin_amdgcn_wmma_f32_16x16x32_bf16(false, a, false, b,
                                                 (short)0, c, false, false);
}
static __device__ __forceinline__ v16bf cat16(v8bf a, v8bf b) {
  return __builtin_shufflevector(a, b, 0,1,2,3,4,5,6,7,8,9,10,11,12,13,14,15);
}
// A 16x32 frag: lane holds row m; elems 0..7 -> k=hi*8+0..7, 8..15 -> k=16+hi*8+0..7
static __device__ __forceinline__ v16bf load_a(const bf16_t* base, int hi) {
  return cat16(*(const v8bf*)(base + hi * 8), *(const v8bf*)(base + 16 + hi * 8));
}
// B 32x16 frag: lane holds col n; elems 0..7 -> k=hi*8+0..7, 8..15 -> k=16+hi*8+0..7
static __device__ __forceinline__ v16bf load_b(const bf16_t* base, int hi) {
  return cat16(*(const v8bf*)(base + hi * 8), *(const v8bf*)(base + 16 + hi * 8));
}
// LDS byte offset of a __shared__ object (AS3 pointer value == LDS offset)
static __device__ __forceinline__ unsigned lds_off(const void* p) {
  return (unsigned)(uintptr_t)(__attribute__((address_space(3))) const void*)p;
}
// 16B global -> LDS async copy (per lane), tracked by ASYNCcnt
static __device__ __forceinline__ void async_b128(unsigned lds_byte, const void* g) {
  asm volatile("global_load_async_to_lds_b128 %0, %1, off"
               :: "v"(lds_byte), "v"(g) : "memory");
}
static __device__ __forceinline__ void wait_async0() {
  asm volatile("s_wait_asynccnt 0" ::: "memory");
}

// ---------------- weight prep ----------------------------------------------
__global__ void k_prep_w(const float* __restrict__ topw, const float* __restrict__ cenw,
                         const float* __restrict__ botw, const float* __restrict__ outw,
                         bf16_t* __restrict__ topw_bf, bf16_t* __restrict__ cenw_bf,
                         bf16_t* __restrict__ botw_p,  bf16_t* __restrict__ outw_p) {
  int idx = blockIdx.x * blockDim.x + threadIdx.x;
  if (idx < P_ * C_) {
    topw_bf[idx] = (bf16_t)topw[idx];
    cenw_bf[idx] = (bf16_t)cenw[idx];
  }
  if (idx < C_ * K9_) {
    int o = idx / K9_, rem = idx % K9_;
    int tap = rem / C_, c = rem % C_;
    botw_p[idx] = (bf16_t)botw[((size_t)o * C_ + c) * 9 + tap];
    outw_p[idx] = (bf16_t)outw[((size_t)o * C_ + c) * 9 + tap];
  }
}

__global__ void k_prep_x(const float* __restrict__ x, bf16_t* __restrict__ xbf, int n) {
  int idx = blockIdx.x * blockDim.x + threadIdx.x;
  if (idx < n) xbf[idx] = (bf16_t)x[idx];
}

// ---------------- 1x1 convs (top & center), fused GEMM ----------------------
// M=P(64), N=128 HW tile, K=C(256). Writes topT and centerT, both (B,HW,P).
__global__ __launch_bounds__(256) void k_gemm1x1(
    const bf16_t* __restrict__ xbf, const bf16_t* __restrict__ wt,
    const bf16_t* __restrict__ wc,  const float* __restrict__ bt,
    const float* __restrict__ bc,   bf16_t* __restrict__ topT,
    bf16_t* __restrict__ cenT) {
  __shared__ bf16_t xsT[128][40];          // transposed tile: [n][k], padded row
  const int tid = threadIdx.x, lane = tid & 31, wv = tid >> 5;
  const int ln = lane & 15, hi = lane >> 4;
  const int b  = blockIdx.x >> 5;
  const int n0 = (blockIdx.x & 31) * 128;
  const int i  = wv >> 1;                  // 4 row blocks (M=64)
  const int j0 = (wv & 1) * 4;             // 8 col blocks (N=128), 4 per wave

  v8f at[4] = {}, ac[4] = {};
  for (int ks = 0; ks < 8; ++ks) {
    const int c0 = ks * 32;
    {   // stage 32x128 x-tile transposed (coalesced loads, scattered stores)
      const int k = tid >> 3, nb = (tid & 7) * 16;
      const bf16_t* src = xbf + ((size_t)b * C_ + (c0 + k)) * HW_ + n0 + nb;
#pragma unroll
      for (int q = 0; q < 16; ++q) xsT[nb + q][k] = src[q];
    }
    __syncthreads();
    const v16bf Atp = load_a(wt + (i * 16 + ln) * C_ + c0, hi);
    const v16bf Acn = load_a(wc + (i * 16 + ln) * C_ + c0, hi);
#pragma unroll
    for (int jj = 0; jj < 4; ++jj) {
      const v16bf Bf = load_b(&xsT[(j0 + jj) * 16 + ln][0], hi);
      at[jj] = wmma_bf16(Atp, Bf, at[jj]);
      ac[jj] = wmma_bf16(Acn, Bf, ac[jj]);
    }
    __syncthreads();
  }
  const int p0 = i * 16 + hi * 8;
#pragma unroll
  for (int jj = 0; jj < 4; ++jj) {
    const int n = n0 + (j0 + jj) * 16 + ln;
    v8bf ot, oc;
#pragma unroll
    for (int r = 0; r < 8; ++r) {
      ot[r] = (bf16_t)(at[jj][r] + bt[p0 + r]);
      oc[r] = (bf16_t)(ac[jj][r] + bc[p0 + r]);
    }
    *(v8bf*)(topT + ((size_t)b * HW_ + n) * P_ + p0) = ot;  // 16B store
    *(v8bf*)(cenT + ((size_t)b * HW_ + n) * P_ + p0) = oc;  // 16B store
  }
}

// ---------------- 3x3 conv as implicit GEMM -------------------------------
// M=256 out ch, N=64 (one image row), K=2304 tap-major. One WG per (b,y).
// STORE_BF=1: bf16 NCHW (attention V). STORE_BF=0: f32 NCHW (final output).
template<int STORE_BF>
__global__ __launch_bounds__(256) void k_conv3x3(
    const bf16_t* __restrict__ in, const bf16_t* __restrict__ wp,
    const float* __restrict__ bias, float* __restrict__ out_f,
    bf16_t* __restrict__ out_b16) {
  __shared__ bf16_t bsT[64][40];           // transposed tile: [x][c], padded
  const int tid = threadIdx.x, lane = tid & 31, wv = tid >> 5;
  const int ln = lane & 15, hi = lane >> 4;
  const int b = blockIdx.x >> 6, y = blockIdx.x & 63;

  v8f acc[2][4] = {};
  for (int step = 0; step < 72; ++step) {
    const int tap = step >> 3;
    const int ky = tap / 3 - 1, kx = tap % 3 - 1;
    const int c0 = (step & 7) * 32;
    {   // stage shifted, zero-padded input row transposed: 64 px x 32 ch
      const int cc = tid >> 3, x0 = (tid & 7) * 8;
      const int iy = y + ky;
#pragma unroll
      for (int q = 0; q < 8; ++q) {
        const int ix = x0 + q + kx;
        bf16_t v = (bf16_t)0.0f;
        if (iy >= 0 && iy < 64 && ix >= 0 && ix < 64)
          v = in[((size_t)b * C_ + (c0 + cc)) * HW_ + iy * 64 + ix];
        bsT[x0 + q][cc] = v;
      }
    }
    __syncthreads();
#pragma unroll
    for (int rr = 0; rr < 2; ++rr) {
      const int rb = wv * 2 + rr;          // 16 row blocks, 2 per wave
      const v16bf Af = load_a(wp + (size_t)(rb * 16 + ln) * K9_ + tap * C_ + c0, hi);
#pragma unroll
      for (int jj = 0; jj < 4; ++jj) {
        const v16bf Bf = load_b(&bsT[jj * 16 + ln][0], hi);
        acc[rr][jj] = wmma_bf16(Af, Bf, acc[rr][jj]);
      }
    }
    __syncthreads();
  }
#pragma unroll
  for (int rr = 0; rr < 2; ++rr) {
    const int rb = wv * 2 + rr;
#pragma unroll
    for (int jj = 0; jj < 4; ++jj) {
      const int n = y * 64 + jj * 16 + ln;
#pragma unroll
      for (int r = 0; r < 8; ++r) {
        const int o = rb * 16 + r + hi * 8;
        const float v = acc[rr][jj][r] + bias[o];
        if (STORE_BF) out_b16[((size_t)b * C_ + o) * HW_ + n] = (bf16_t)v;
        else          out_f  [((size_t)b * C_ + o) * HW_ + n] = v;
      }
    }
  }
}

// Stage one contiguous 64x64 bf16 tile (8KB) into padded LDS rows [64][72]
// using async b128 copies: 512 chunks of 16B, 2 per thread.
static __device__ __forceinline__ void stage_tile_async(
    const bf16_t* tile, unsigned lds_base, int tid) {
#pragma unroll
  for (int q = tid; q < 512; q += 256) {
    const unsigned lo = lds_base + (unsigned)(q >> 3) * (72 * 2) + (unsigned)(q & 7) * 16;
    async_b128(lo, (const char*)tile + q * 16);
  }
  wait_async0();
}

// ---------------- pass A: S tiles + online (max, sumexp) per WG -------------
__global__ __launch_bounds__(256) void k_attn_stats(
    const bf16_t* __restrict__ cenT, const bf16_t* __restrict__ topT,
    float* __restrict__ stats) {
  __shared__ bf16_t topsT[64][72];         // [m][p], padded row (144B, 16B-aligned)
  __shared__ float  Sst[64][64];
  __shared__ float  red[256];
  __shared__ float  runm, runs;
  const int tid = threadIdx.x, lane = tid & 31, wv = tid >> 5;
  const int ln = lane & 15, hi = lane >> 4;
  const int b = blockIdx.x >> 6;
  const int n0 = (blockIdx.x & 63) * 64;
  const int i = wv >> 1, j0 = (wv & 1) * 2;
  const unsigned tops_base = lds_off(&topsT[0][0]);

  v16bf Ac[2];                             // centerT A-frags, invariant over m
#pragma unroll
  for (int ks = 0; ks < 2; ++ks)
    Ac[ks] = load_a(cenT + ((size_t)b * HW_ + n0 + i * 16 + ln) * P_ + ks * 32, hi);
  if (tid == 0) { runm = -INFINITY; runs = 0.0f; }
  __syncthreads();

  for (int mt = 0; mt < 64; ++mt) {
    const int m0 = mt * 64;
    const bf16_t* tile = topT + ((size_t)b * HW_ + m0) * P_;   // contiguous 8KB
    stage_tile_async(tile, tops_base, tid);
    if (mt + 1 < 64) __builtin_prefetch((const char*)tile + 8192 + tid * 32, 0, 0);
    __syncthreads();
#pragma unroll
    for (int jj = 0; jj < 2; ++jj) {
      const int j = j0 + jj;
      v8f acc = {};
#pragma unroll
      for (int ks = 0; ks < 2; ++ks)
        acc = wmma_bf16(Ac[ks], load_b(&topsT[j * 16 + ln][ks * 32], hi), acc);
#pragma unroll
      for (int r = 0; r < 8; ++r) Sst[i * 16 + r + hi * 8][j * 16 + ln] = acc[r];
    }
    __syncthreads();
    // online softmax statistics over the 4096 logits of this tile
    const float* sf = &Sst[0][0];
    float lm = -INFINITY;
#pragma unroll
    for (int q = 0; q < 16; ++q) lm = fmaxf(lm, sf[tid * 16 + q]);
    red[tid] = lm; __syncthreads();
    for (int s = 128; s > 0; s >>= 1) { if (tid < s) red[tid] = fmaxf(red[tid], red[tid + s]); __syncthreads(); }
    const float nm = fmaxf(runm, red[0]);
    __syncthreads();
    float ls = 0.0f;
#pragma unroll
    for (int q = 0; q < 16; ++q) ls += __expf(sf[tid * 16 + q] - nm);
    red[tid] = ls; __syncthreads();
    for (int s = 128; s > 0; s >>= 1) { if (tid < s) red[tid] += red[tid + s]; __syncthreads(); }
    if (tid == 0) { runs = runs * __expf(runm - nm) + red[0]; runm = nm; }
    __syncthreads();
  }
  if (tid == 0) { stats[blockIdx.x * 2] = runm; stats[blockIdx.x * 2 + 1] = runs; }
}

// ---------------- tiny per-batch reduction of (max,sum) pairs ---------------
__global__ __launch_bounds__(64) void k_reduce_stats(const float* __restrict__ stats,
                                                     float* __restrict__ MZ) {
  __shared__ float rm[64], rs[64];
  const int b = blockIdx.x, t = threadIdx.x;
  const float m = stats[(b * 64 + t) * 2], s = stats[(b * 64 + t) * 2 + 1];
  rm[t] = m; __syncthreads();
  for (int k = 32; k > 0; k >>= 1) { if (t < k) rm[t] = fmaxf(rm[t], rm[t + k]); __syncthreads(); }
  const float M = rm[0]; __syncthreads();
  rs[t] = s * __expf(m - M); __syncthreads();
  for (int k = 32; k > 0; k >>= 1) { if (t < k) rs[t] += rs[t + k]; __syncthreads(); }
  if (t == 0) { MZ[b * 2] = M; MZ[b * 2 + 1] = rs[0]; }
}

// ---------------- pass B: recompute S, O = exp(S-M)/Z @ V -------------------
__global__ __launch_bounds__(256) void k_attn_apply(
    const bf16_t* __restrict__ cenT, const bf16_t* __restrict__ topT,
    const bf16_t* __restrict__ v,    const float* __restrict__ MZ,
    float* __restrict__ ybuf) {
  __shared__ bf16_t topsT[64][72];
  __shared__ float  Sst[64][64];
  __shared__ bf16_t Pst[64][72];           // [n][m], padded row
  const int tid = threadIdx.x, lane = tid & 31, wv = tid >> 5;
  const int ln = lane & 15, hi = lane >> 4;
  const int b = blockIdx.x >> 6;
  const int n0 = (blockIdx.x & 63) * 64;
  const int i = wv >> 1, j0 = (wv & 1) * 2;
  const int cb0 = (wv & 1) * 8;            // 16 c-blocks (C=256), 8 per wave
  const float M = MZ[b * 2];
  const float invZ = 1.0f / MZ[b * 2 + 1];
  const unsigned tops_base = lds_off(&topsT[0][0]);

  v16bf Ac[2];
#pragma unroll
  for (int ks = 0; ks < 2; ++ks)
    Ac[ks] = load_a(cenT + ((size_t)b * HW_ + n0 + i * 16 + ln) * P_ + ks * 32, hi);
  v8f Oa[8] = {};

  for (int mt = 0; mt < 64; ++mt) {
    const int m0 = mt * 64;
    const bf16_t* tile = topT + ((size_t)b * HW_ + m0) * P_;
    stage_tile_async(tile, tops_base, tid);
    if (mt + 1 < 64) __builtin_prefetch((const char*)tile + 8192 + tid * 32, 0, 0);
    __syncthreads();
#pragma unroll
    for (int jj = 0; jj < 2; ++jj) {
      const int j = j0 + jj;
      v8f acc = {};
#pragma unroll
      for (int ks = 0; ks < 2; ++ks)
        acc = wmma_bf16(Ac[ks], load_b(&topsT[j * 16 + ln][ks * 32], hi), acc);
#pragma unroll
      for (int r = 0; r < 8; ++r) Sst[i * 16 + r + hi * 8][j * 16 + ln] = acc[r];
    }
    __syncthreads();
    {   // P = exp(S - M) in bf16; each thread owns 16 contiguous elems of a row
      const int row = tid >> 2, cl = (tid & 3) * 16;
      v8bf p0, p1;
#pragma unroll
      for (int q = 0; q < 8; ++q) p0[q] = (bf16_t)__expf(Sst[row][cl + q] - M);
#pragma unroll
      for (int q = 0; q < 8; ++q) p1[q] = (bf16_t)__expf(Sst[row][cl + 8 + q] - M);
      *(v8bf*)(&Pst[row][cl])     = p0;
      *(v8bf*)(&Pst[row][cl + 8]) = p1;
    }
    __syncthreads();
    // O(64x256) += P(64x64) @ V(64x256); V in natural (B,C,HW) -> B-frags are
    // two contiguous global_load_b128 per lane (k = m contiguous).
#pragma unroll
    for (int ks = 0; ks < 2; ++ks) {
      const v16bf Ap = load_a(&Pst[i * 16 + ln][ks * 32], hi);
#pragma unroll
      for (int cb = 0; cb < 8; ++cb) {
        const int c = (cb0 + cb) * 16 + ln;
        const v16bf Bf = load_b(v + ((size_t)b * C_ + c) * HW_ + m0 + ks * 32, hi);
        Oa[cb] = wmma_bf16(Ap, Bf, Oa[cb]);
      }
    }
    __syncthreads();
  }
#pragma unroll
  for (int cb = 0; cb < 8; ++cb) {
    const int c = (cb0 + cb) * 16 + ln;
#pragma unroll
    for (int r = 0; r < 8; ++r) {
      const int n = n0 + i * 16 + r + hi * 8;
      ybuf[((size_t)b * HW_ + n) * C_ + c] = Oa[cb][r] * invZ;
    }
  }
}

// y = x + reshape(O): the reference reshapes (B,HW,C)->(B,C,H,W) WITHOUT
// transpose, so this is a flat elementwise add per batch. Emit bf16 for conv.
__global__ void k_residual(const float* __restrict__ x, const float* __restrict__ ybuf,
                           bf16_t* __restrict__ ybf, int n) {
  int idx = blockIdx.x * blockDim.x + threadIdx.x;
  if (idx < n) ybf[idx] = (bf16_t)(x[idx] + ybuf[idx]);
}

// ---------------------------------------------------------------------------
extern "C" void kernel_launch(void* const* d_in, const int* in_sizes, int n_in,
                              void* d_out, int out_size, void* d_ws, size_t ws_size,
                              hipStream_t stream) {
  const float* x     = (const float*)d_in[0];
  const float* top_w = (const float*)d_in[1];
  const float* top_b = (const float*)d_in[2];
  const float* cen_w = (const float*)d_in[3];
  const float* cen_b = (const float*)d_in[4];
  const float* bot_w = (const float*)d_in[5];
  const float* bot_b = (const float*)d_in[6];
  const float* out_w = (const float*)d_in[7];
  const float* out_b = (const float*)d_in[8];
  float* out = (float*)d_out;

  char* ws = (char*)d_ws;
  size_t off = 0;
  auto alloc = [&](size_t bytes) -> void* {
    void* p = ws + off;
    off = (off + bytes + 255) & ~(size_t)255;
    return p;
  };
  bf16_t* xbf     = (bf16_t*)alloc((size_t)B_ * C_ * HW_ * 2);
  bf16_t* topw_bf = (bf16_t*)alloc((size_t)P_ * C_ * 2);
  bf16_t* cenw_bf = (bf16_t*)alloc((size_t)P_ * C_ * 2);
  bf16_t* botw_p  = (bf16_t*)alloc((size_t)C_ * K9_ * 2);
  bf16_t* outw_p  = (bf16_t*)alloc((size_t)C_ * K9_ * 2);
  bf16_t* topT    = (bf16_t*)alloc((size_t)B_ * HW_ * P_ * 2);
  bf16_t* cenT    = (bf16_t*)alloc((size_t)B_ * HW_ * P_ * 2);
  bf16_t* vbuf    = (bf16_t*)alloc((size_t)B_ * C_ * HW_ * 2);
  float*  ybuf    = (float*) alloc((size_t)B_ * HW_ * C_ * 4);
  bf16_t* ybf     = (bf16_t*)alloc((size_t)B_ * C_ * HW_ * 2);
  float*  stats   = (float*) alloc((size_t)B_ * 64 * 2 * 4);
  float*  MZ      = (float*) alloc((size_t)B_ * 2 * 4);
  if (ws_size < off) return;   // ~95 MB of scratch required

  const int nx = B_ * C_ * HW_;
  k_prep_w<<<(C_ * K9_ + 255) / 256, 256, 0, stream>>>(
      top_w, cen_w, bot_w, out_w, topw_bf, cenw_bf, botw_p, outw_p);
  k_prep_x<<<(nx + 255) / 256, 256, 0, stream>>>(x, xbf, nx);
  k_gemm1x1<<<B_ * (HW_ / 128), 256, 0, stream>>>(
      xbf, topw_bf, cenw_bf, top_b, cen_b, topT, cenT);
  k_conv3x3<1><<<B_ * 64, 256, 0, stream>>>(xbf, botw_p, bot_b, nullptr, vbuf);
  k_attn_stats<<<B_ * 64, 256, 0, stream>>>(cenT, topT, stats);
  k_reduce_stats<<<B_, 64, 0, stream>>>(stats, MZ);
  k_attn_apply<<<B_ * 64, 256, 0, stream>>>(cenT, topT, vbuf, MZ, ybuf);
  k_residual<<<(nx + 255) / 256, 256, 0, stream>>>(x, ybuf, ybf, nx);
  k_conv3x3<0><<<B_ * 64, 256, 0, stream>>>(ybf, outw_p, out_b, out, nullptr);
}
